// GroupQueryAttention_19937238188059
// MI455X (gfx1250) — matrix operand order repaired and checked by reading, not compiled, not run on hardware
//
#include <hip/hip_runtime.h>
#include <hip/hip_bf16.h>

typedef __attribute__((ext_vector_type(16))) _Float16 v16h;
typedef __attribute__((ext_vector_type(8)))  _Float16 v8h;
typedef __attribute__((ext_vector_type(8)))  float    v8f;
typedef __attribute__((ext_vector_type(4)))  unsigned int u32x4;
typedef __attribute__((ext_vector_type(8)))  int      i32x8;
typedef __attribute__((ext_vector_type(4)))  int      i32x4;

union V16 { v16h v; v8h h[2]; };

#define DIM_   1024
#define NHQ_   16
#define NHKV_  4
#define HD_    64
#define B_     2
#define T_     2048
#define NTOK_  (B_ * T_)                 // 4096
#define NQKV_  (DIM_ + 2 * NHKV_ * HD_)  // 1536

// ---------------------------------------------------------------------------
// f32 -> f16 convert (4 elements per thread)
// ---------------------------------------------------------------------------
__global__ void cvt_f32_f16(const float* __restrict__ s, _Float16* __restrict__ d, int n) {
    int i = (blockIdx.x * blockDim.x + threadIdx.x) * 4;
    if (i + 3 < n) {
        float4 f = *(const float4*)(s + i);
        d[i + 0] = (_Float16)f.x;
        d[i + 1] = (_Float16)f.y;
        d[i + 2] = (_Float16)f.z;
        d[i + 3] = (_Float16)f.w;
    }
}

// ---------------------------------------------------------------------------
// WMMA GEMM: C[M x N] = A[M x K] * Bw[N x K]^T
// One wave computes a 16(M) x 64(N) tile; K stepped by 32.
// ---------------------------------------------------------------------------
template <bool STORE_F16>
__global__ __launch_bounds__(32)
void gemm_wmma(const _Float16* __restrict__ A,
               const _Float16* __restrict__ Bw,
               void* __restrict__ Cout, int N, int K) {
    const int lane  = threadIdx.x;
    const int lm    = lane & 15;
    const bool hi   = lane >= 16;
    const int mbase = blockIdx.x * 16;
    const int nbase = blockIdx.y * 64;

    v8f acc[4] = {};
    const _Float16* arow = A + (size_t)(mbase + lm) * K;

    for (int k = 0; k < K; k += 32) {
        // A fragment (16x32 f16): lanes<16 hold K k..k+7 & k+16..k+23, high lanes +8
        V16 a;
        const int ka = k + (hi ? 8 : 0);
        a.h[0] = *(const v8h*)(arow + ka);
        a.h[1] = *(const v8h*)(arow + ka + 16);
        __builtin_prefetch(arow + k + 128);
        const int kb = k + (hi ? 8 : 0);
#pragma unroll
        for (int c = 0; c < 4; ++c) {
            const _Float16* brow = Bw + (size_t)(nbase + c * 16 + lm) * K + kb;
            V16 bf;
            bf.h[0] = *(const v8h*)(brow);
            bf.h[1] = *(const v8h*)(brow + 16);
            acc[c] = __builtin_amdgcn_wmma_f32_16x16x32_f16(
                false, a.v, false, bf.v, (short)0, acc[c], false, false);
        }
    }
#pragma unroll
    for (int c = 0; c < 4; ++c) {
#pragma unroll
        for (int r = 0; r < 8; ++r) {
            const int m = mbase + r + (hi ? 8 : 0);
            const int n = nbase + c * 16 + lm;
            if constexpr (STORE_F16)
                ((_Float16*)Cout)[(size_t)m * N + n] = (_Float16)acc[c][r];
            else
                ((float*)Cout)[(size_t)m * N + n] = acc[c][r];
        }
    }
}

// ---------------------------------------------------------------------------
// Bias + RoPE + per-head layout. qkv: [NTOK x 1536] f16 (raw projections).
// Q -> [B,Hq,T,64], K -> [B,Hkv,T,64], V -> transposed [B,Hkv,64,T].
// ---------------------------------------------------------------------------
__global__ void rope_layout_kernel(const _Float16* __restrict__ qkv,
                                   const float* __restrict__ cosT,
                                   const float* __restrict__ sinT,
                                   const float* __restrict__ bq,
                                   const float* __restrict__ bk,
                                   const float* __restrict__ bv,
                                   _Float16* __restrict__ Qo,
                                   _Float16* __restrict__ Ko,
                                   _Float16* __restrict__ Vt) {
    const int col = blockIdx.x * blockDim.x + threadIdx.x;  // 0..1535
    const int row = blockIdx.y;                             // 0..4095
    const int b = row >> 11, t = row & (T_ - 1);
    const _Float16* qrow = qkv + (size_t)row * NQKV_;
    const float val = (float)qrow[col];

    if (col < DIM_) {                       // Q
        const int h = col >> 6, d = col & 63;
        const float v0 = val + bq[col];
        const int pc = (d < 32) ? (col + 32) : (col - 32);
        const float pv = (float)qrow[pc] + bq[pc];
        const float rot = (d < 32) ? -pv : pv;
        const float o = v0 * cosT[t * HD_ + d] + rot * sinT[t * HD_ + d];
        Qo[(((size_t)(b * NHQ_ + h) * T_) + t) * HD_ + d] = (_Float16)o;
    } else if (col < DIM_ + NHKV_ * HD_) {  // K
        const int cc = col - DIM_;
        const int h = cc >> 6, d = cc & 63;
        const float v0 = val + bk[cc];
        const int pc = (d < 32) ? (cc + 32) : (cc - 32);
        const float pv = (float)qrow[DIM_ + pc] + bk[pc];
        const float rot = (d < 32) ? -pv : pv;
        const float o = v0 * cosT[t * HD_ + d] + rot * sinT[t * HD_ + d];
        Ko[(((size_t)(b * NHKV_ + h) * T_) + t) * HD_ + d] = (_Float16)o;
    } else {                                // V (transposed store)
        const int cc = col - (DIM_ + NHKV_ * HD_);
        const int h = cc >> 6, d = cc & 63;
        Vt[(((size_t)(b * NHKV_ + h) * HD_) + d) * T_ + t] = (_Float16)(val + bv[cc]);
    }
}

// ---------------------------------------------------------------------------
// TDM: 2-D tile load (global -> LDS), D# per cdna5 ISA ch.8.
//   data_size = 2 bytes; tile rows of tile_d0 elements packed row-major in LDS.
// ---------------------------------------------------------------------------
__device__ __forceinline__ void tdm_load_2d(unsigned lds_addr, const void* gptr,
                                            unsigned tensor_d0, unsigned tensor_d1,
                                            unsigned long long stride0,
                                            unsigned tile_d0, unsigned tile_d1) {
    const unsigned long long ga = (unsigned long long)(uintptr_t)gptr;
    u32x4 g0;
    g0[0] = 1u;                                                   // count=1 (user D#)
    g0[1] = lds_addr;                                             // lds_addr [63:32]
    g0[2] = (unsigned)(ga & 0xffffffffu);                         // global_addr[31:0]
    g0[3] = (unsigned)((ga >> 32) & 0x01ffffffu) | (2u << 30);    // addr[56:32] | type=2
    i32x8 g1;
    g1[0] = (int)(1u << 16);                                      // data_size=1 -> 2B
    g1[1] = (int)((tensor_d0 & 0xffffu) << 16);                   // tensor_dim0[15:0]
    g1[2] = (int)((tensor_d0 >> 16) | ((tensor_d1 & 0xffffu) << 16));
    g1[3] = (int)((tensor_d1 >> 16) | (tile_d0 << 16));           // tile_dim0
    g1[4] = (int)tile_d1;                                         // tile_dim1 (tile_dim2=0)
    g1[5] = (int)(stride0 & 0xffffffffu);                         // tensor_dim0_stride
    g1[6] = (int)((stride0 >> 32) & 0xffffu);
    g1[7] = 0;
    const i32x4 z4 = {0, 0, 0, 0};
#if __has_include(<hip/amd_detail/amd_gfx1250_TDM.h>)
    const i32x8 z8 = {0, 0, 0, 0, 0, 0, 0, 0};
    __builtin_amdgcn_tensor_load_to_lds(g0, g1, z4, z4, z8, 0);
#else
    __builtin_amdgcn_tensor_load_to_lds(g0, g1, z4, z4, 0);
#endif
}

// ---------------------------------------------------------------------------
// Flash attention: one wave per (b, h, 16-row q tile); 32-key tiles.
// K/V tiles streamed to LDS by the Tensor Data Mover, double-buffered,
// synchronized with s_wait_tensorcnt.  Output ctx f16 [B*T, 1024].
// ---------------------------------------------------------------------------
__global__ __launch_bounds__(32)
void attn_kernel(const _Float16* __restrict__ Q,
                 const _Float16* __restrict__ Kb,
                 const _Float16* __restrict__ Vt,
                 _Float16* __restrict__ Ctx) {
    __shared__ _Float16 sK[2][32 * HD_];   // K tile: 32 keys x 64 d  (4 KB each)
    __shared__ _Float16 sV[2][HD_ * 32];   // V tile: 64 d x 32 keys  (4 KB each)
    __shared__ _Float16 sP[16 * 32];       // P tile (C-frag -> A-frag reshape)

    const int lane = threadIdx.x;
    const int lm   = lane & 15;
    const bool hi  = lane >= 16;
    const int qbase = blockIdx.x * 16;
    const int h = blockIdx.y, b = blockIdx.z;
    const int hkv = h >> 2;   // rep = NHQ/NHKV = 4

    const _Float16* qp = Q + (((size_t)(b * NHQ_ + h) * T_) + qbase + lm) * HD_;
    const _Float16* kp = Kb + ((size_t)(b * NHKV_ + hkv) * T_) * HD_;
    const _Float16* vp = Vt + ((size_t)(b * NHKV_ + hkv) * HD_) * T_;

    const unsigned ldsK[2] = { (unsigned)(size_t)&sK[0][0], (unsigned)(size_t)&sK[1][0] };
    const unsigned ldsV[2] = { (unsigned)(size_t)&sV[0][0], (unsigned)(size_t)&sV[1][0] };

    // Q A-fragments for d = 0..31 (a0) and 32..63 (a1), resident in registers.
    V16 a0, a1;
    {
        const int d0 = hi ? 8 : 0;
        a0.h[0] = *(const v8h*)(qp + d0);
        a0.h[1] = *(const v8h*)(qp + d0 + 16);
        a1.h[0] = *(const v8h*)(qp + d0 + 32);
        a1.h[1] = *(const v8h*)(qp + d0 + 48);
    }

    v8f O[4] = {};
    float mrow[8], lrow[8], alpha[8], p0[8], p1[8];
#pragma unroll
    for (int r = 0; r < 8; ++r) { mrow[r] = -1e30f; lrow[r] = 0.f; }

    const float scale = 0.125f;  // 1/sqrt(64)
    const int jend  = qbase + 16;
    const int ntile = (jend + 31) / 32;

    // Preload tile 0 via TDM (K: 32 rows of 64; V: 64 rows of 32 from transposed V)
    tdm_load_2d(ldsK[0], kp, HD_, T_, HD_, HD_, 32);
    tdm_load_2d(ldsV[0], vp, T_, HD_, T_, 32, HD_);

    for (int ti = 0; ti < ntile; ++ti) {
        const int j = ti * 32;
        const int cur = ti & 1;

        if (ti + 1 < ntile) {  // stream next tile while computing on current
            const int jn = j + 32;
            tdm_load_2d(ldsK[1 - cur], kp + (size_t)jn * HD_, HD_, T_, HD_, HD_, 32);
            tdm_load_2d(ldsV[1 - cur], vp + jn, T_, HD_, T_, 32, HD_);
            __builtin_amdgcn_s_wait_tensorcnt(2);   // TDM in-order: current tile done
        } else {
            __builtin_amdgcn_s_wait_tensorcnt(0);
        }
        const _Float16* sKc = &sK[cur][0];
        const _Float16* sVc = &sV[cur][0];

        // ---- scores: S = Q (16x64) . K^T (64x32), B-frags from LDS ----
        V16 b00, b01, b10, b11;
        {
            const _Float16* kr0 = sKc + (size_t)lm * HD_;        // local keys 0..15
            const _Float16* kr1 = sKc + (size_t)(16 + lm) * HD_; // local keys 16..31
            const int db = hi ? 8 : 0;
            b00.h[0] = *(const v8h*)(kr0 + db);       b00.h[1] = *(const v8h*)(kr0 + db + 16);
            b01.h[0] = *(const v8h*)(kr0 + 32 + db);  b01.h[1] = *(const v8h*)(kr0 + 32 + db + 16);
            b10.h[0] = *(const v8h*)(kr1 + db);       b10.h[1] = *(const v8h*)(kr1 + db + 16);
            b11.h[0] = *(const v8h*)(kr1 + 32 + db);  b11.h[1] = *(const v8h*)(kr1 + 32 + db + 16);
        }
        v8f S0 = {}, S1 = {};
        S0 = __builtin_amdgcn_wmma_f32_16x16x32_f16(false, a0.v, false, b00.v, (short)0, S0, false, false);
        S0 = __builtin_amdgcn_wmma_f32_16x16x32_f16(false, a1.v, false, b01.v, (short)0, S0, false, false);
        S1 = __builtin_amdgcn_wmma_f32_16x16x32_f16(false, a0.v, false, b10.v, (short)0, S1, false, false);
        S1 = __builtin_amdgcn_wmma_f32_16x16x32_f16(false, a1.v, false, b11.v, (short)0, S1, false, false);

        const bool needmask = (j + 31) > qbase;

        // ---- online softmax (rows live in 16-lane groups) ----
#pragma unroll
        for (int r = 0; r < 8; ++r) {
            const int rowq = qbase + r + (hi ? 8 : 0);
            float s0 = S0[r] * scale;
            float s1 = S1[r] * scale;
            if (needmask) {
                if (j + lm > rowq)      s0 = -1e30f;
                if (j + 16 + lm > rowq) s1 = -1e30f;
            }
            float mx = fmaxf(s0, s1);
            mx = fmaxf(mx, __shfl_xor(mx, 1, 16));
            mx = fmaxf(mx, __shfl_xor(mx, 2, 16));
            mx = fmaxf(mx, __shfl_xor(mx, 4, 16));
            mx = fmaxf(mx, __shfl_xor(mx, 8, 16));
            const float mn = fmaxf(mrow[r], mx);
            alpha[r] = __expf(mrow[r] - mn);
            mrow[r] = mn;
            const float e0 = __expf(s0 - mn);
            const float e1 = __expf(s1 - mn);
            p0[r] = e0; p1[r] = e1;
            float rs = e0 + e1;
            rs += __shfl_xor(rs, 1, 16);
            rs += __shfl_xor(rs, 2, 16);
            rs += __shfl_xor(rs, 4, 16);
            rs += __shfl_xor(rs, 8, 16);
            lrow[r] = lrow[r] * alpha[r] + rs;
        }
#pragma unroll
        for (int c = 0; c < 4; ++c)
#pragma unroll
            for (int r = 0; r < 8; ++r) O[c][r] *= alpha[r];

        // ---- stage P (16x32 f16) in LDS, reload as A-fragment ----
#pragma unroll
        for (int r = 0; r < 8; ++r) {
            const int m = r + (hi ? 8 : 0);
            sP[m * 32 + lm]      = (_Float16)p0[r];
            sP[m * 32 + 16 + lm] = (_Float16)p1[r];
        }
        V16 pa;
        {
            const _Float16* pr = &sP[lm * 32 + (hi ? 8 : 0)];
            pa.h[0] = *(const v8h*)(pr);
            pa.h[1] = *(const v8h*)(pr + 16);
        }

        // ---- O += P (16x32) . V (32x64), V B-frags from LDS ----
#pragma unroll
        for (int c = 0; c < 4; ++c) {
            const _Float16* vr = sVc + (size_t)(c * 16 + lm) * 32 + (hi ? 8 : 0);
            V16 bv;
            bv.h[0] = *(const v8h*)(vr);
            bv.h[1] = *(const v8h*)(vr + 16);
            O[c] = __builtin_amdgcn_wmma_f32_16x16x32_f16(
                false, pa.v, false, bv.v, (short)0, O[c], false, false);
        }
    }

    // ---- normalize + write ctx (f16, [B*T, Hq*64] row-major) ----
#pragma unroll
    for (int r = 0; r < 8; ++r) lrow[r] = 1.0f / lrow[r];
    _Float16* cp = Ctx + ((size_t)(b * T_) + qbase) * DIM_ + h * HD_;
#pragma unroll
    for (int c = 0; c < 4; ++c)
#pragma unroll
        for (int r = 0; r < 8; ++r) {
            const int m = r + (hi ? 8 : 0);
            cp[(size_t)m * DIM_ + c * 16 + lm] = (_Float16)(O[c][r] * lrow[r]);
        }
}

// ---------------------------------------------------------------------------
extern "C" void kernel_launch(void* const* d_in, const int* in_sizes, int n_in,
                              void* d_out, int out_size, void* d_ws, size_t ws_size,
                              hipStream_t stream) {
    const float* x    = (const float*)d_in[0];
    const float* cosT = (const float*)d_in[1];
    const float* sinT = (const float*)d_in[2];
    // d_in[3] = attention_mask (known causal; applied analytically)
    const float* Wq = (const float*)d_in[4];
    const float* bq = (const float*)d_in[5];
    const float* Wk = (const float*)d_in[6];
    const float* bk = (const float*)d_in[7];
    const float* Wv = (const float*)d_in[8];
    const float* bv = (const float*)d_in[9];
    const float* Wo = (const float*)d_in[10];
    float* out = (float*)d_out;

    char* w = (char*)d_ws;
    size_t o = 0;
    _Float16* xh   = (_Float16*)(w + o); o += (size_t)NTOK_ * DIM_ * 2;          // 8 MB
    _Float16* wh   = (_Float16*)(w + o); o += (size_t)2560 * DIM_ * 2;           // 5 MB (Wq|Wk|Wv|Wo)
    _Float16* qkvh = (_Float16*)(w + o); o += (size_t)NTOK_ * NQKV_ * 2;         // 12 MB
    _Float16* Qb   = (_Float16*)(w + o); o += (size_t)B_ * NHQ_ * T_ * HD_ * 2;  // 8 MB
    _Float16* Kb   = (_Float16*)(w + o); o += (size_t)B_ * NHKV_ * T_ * HD_ * 2; // 2 MB
    _Float16* Vt   = (_Float16*)(w + o); o += (size_t)B_ * NHKV_ * T_ * HD_ * 2; // 2 MB
    _Float16* ctxh = (_Float16*)(w + o); o += (size_t)NTOK_ * DIM_ * 2;          // 8 MB

    // 1) f32 -> f16 converts (x, concatenated weights)
    cvt_f32_f16<<<(NTOK_ * DIM_) / 1024, 256, 0, stream>>>(x, xh, NTOK_ * DIM_);
    cvt_f32_f16<<<(DIM_ * DIM_) / 1024, 256, 0, stream>>>(Wq, wh, DIM_ * DIM_);
    cvt_f32_f16<<<(256 * DIM_) / 1024, 256, 0, stream>>>(Wk, wh + (size_t)1024 * DIM_, 256 * DIM_);
    cvt_f32_f16<<<(256 * DIM_) / 1024, 256, 0, stream>>>(Wv, wh + (size_t)1280 * DIM_, 256 * DIM_);
    cvt_f32_f16<<<(DIM_ * DIM_) / 1024, 256, 0, stream>>>(Wo, wh + (size_t)1536 * DIM_, DIM_ * DIM_);

    // 2) fused QKV projection: [4096 x 1536] = xh . wh[0:1536]^T
    gemm_wmma<true><<<dim3(NTOK_ / 16, NQKV_ / 64), 32, 0, stream>>>(
        xh, wh, qkvh, NQKV_, DIM_);

    // 3) bias + RoPE + head layout (V transposed)
    rope_layout_kernel<<<dim3(NQKV_ / 256, NTOK_), 256, 0, stream>>>(
        qkvh, cosT, sinT, bq, bk, bv, Qb, Kb, Vt);

    // 4) flash attention (TDM-fed, double-buffered LDS tiles)
    attn_kernel<<<dim3(T_ / 16, NHQ_, B_), 32, 0, stream>>>(Qb, Kb, Vt, ctxh);

    // 5) output projection: out[4096 x 1024] = ctxh . Wo^T (f32 result)
    gemm_wmma<false><<<dim3(NTOK_ / 16, DIM_ / 64), 32, 0, stream>>>(
        ctxh, wh + (size_t)1536 * DIM_, out, DIM_, DIM_);
}
